// DeepSeekMoE_40750649704890
// MI455X (gfx1250) — matrix operand order repaired and checked by reading, not compiled, not run on hardware
//
#include <hip/hip_runtime.h>
#include <hip/hip_bf16.h>

// ---------------------------------------------------------------------------
// DeepSeekMoE on MI455X (gfx1250): compacted top-K dispatch + bf16 WMMA GEMMs
// with Tensor-Data-Mover (TDM) tile loads into LDS.
// ---------------------------------------------------------------------------

#define NTOK 4096   // B*L
#define DDIM 1024   // model dim
#define HDIM 1024   // expert hidden dim
#define NEXP 15     // routed experts
#define KTOP 3      // active experts per token

typedef __attribute__((ext_vector_type(16))) __bf16       v16bf;
typedef __attribute__((ext_vector_type(8)))  float        v8f;
typedef __attribute__((ext_vector_type(4)))  unsigned int v4u;
typedef __attribute__((ext_vector_type(4)))  unsigned int u32x4;
typedef __attribute__((ext_vector_type(4)))  int          i32x4;
typedef __attribute__((ext_vector_type(8)))  int          i32x8;

#if __has_builtin(__builtin_amdgcn_tensor_load_to_lds)
#define HAS_TDM 1
#else
#define HAS_TDM 0
#endif

union FragBF {
    v16bf v;
    v4u   q[2];
};

__device__ __forceinline__ unsigned short f2bf(float f) {
    unsigned int u = __float_as_uint(f);
    unsigned int r = (u + 0x7FFFu + ((u >> 16) & 1u)) >> 16;   // RNE
    return (unsigned short)r;
}
__device__ __forceinline__ float bf2f(unsigned short h) {
    return __uint_as_float(((unsigned int)h) << 16);
}

__device__ __forceinline__ void wait_tensorcnt0() {
#if __has_builtin(__builtin_amdgcn_s_wait_tensorcnt)
    __builtin_amdgcn_s_wait_tensorcnt(0);
#else
    asm volatile("s_wait_tensorcnt 0x0" ::: "memory");
#endif
}

// ---------------------------------------------------------------------------
// TDM: DMA a 128-row x 32-col bf16 tile (row stride 1024 elems) from global
// into LDS laid out as [row][32 + 8 pad] ushorts (pad via D# pad fields:
// interval = 16 DWORDs (code 3), amount = 4 DWORDs (code 3)).
// ---------------------------------------------------------------------------
__device__ __forceinline__ void tdm_load_tile(unsigned lds_off,
                                              const unsigned short* gptr) {
#if HAS_TDM
    unsigned long long ga = (unsigned long long)(size_t)gptr;
    u32x4 g0;
    g0[0] = 1u;                                          // count=1, load-descriptor
    g0[1] = lds_off;                                     // LDS byte address
    g0[2] = (unsigned)(ga & 0xFFFFFFFFu);                // global_addr[31:0]
    g0[3] = (unsigned)((ga >> 32) & 0x01FFFFFFu)         // global_addr[56:32]
            | (2u << 30);                                // type = 2 (image)
    i32x8 g1;
    g1[0] = (int)((1u << 16)                             // data_size = 2 bytes
                | (1u << 20)                             // pad_enable
                | (3u << 22)                             // pad_interval: 16 DWORDs
                | (3u << 25));                           // pad_amount: 4 DWORDs
    unsigned td0 = 1024u;                                // tensor_dim0
    unsigned td1 = 1u << 20;                             // tensor_dim1 (large)
    g1[1] = (int)((td0 & 0xFFFFu) << 16);                // [63:48] = td0 lo16
    g1[2] = (int)((td0 >> 16) | ((td1 & 0xFFFFu) << 16));
    g1[3] = (int)((td1 >> 16) | (32u << 16));            // tile_dim0 = 32
    g1[4] = (int)(128u);                                 // tile_dim1 = 128, tile_dim2 = 0
    g1[5] = (int)(1024u);                                // tensor_dim0_stride lo32
    g1[6] = 0;                                           // stride hi, dim1_stride lo
    g1[7] = 0;
    i32x4 z4; z4[0] = 0; z4[1] = 0; z4[2] = 0; z4[3] = 0;
#if __clang_major__ >= 23
    i32x8 z8;
#pragma unroll
    for (int i = 0; i < 8; ++i) z8[i] = 0;
    __builtin_amdgcn_tensor_load_to_lds(g0, g1, z4, z4, z8, 0);
#else
    __builtin_amdgcn_tensor_load_to_lds(g0, g1, z4, z4, 0);
#endif
#endif
}

// ---------------------------------------------------------------------------
// Workspace layout (bytes)
// ---------------------------------------------------------------------------
constexpr size_t OFF_CNT  = 0;                                         // 15 ints
constexpr size_t OFF_LIST = 256;                                       // 15*4096 ints
constexpr size_t OFF_TKW  = OFF_LIST + (size_t)NEXP * NTOK * 4;        // N*3 floats
constexpr size_t OFF_HS   = OFF_TKW + (size_t)NTOK * KTOP * 4;         // N*H bf16
constexpr size_t OFF_HE   = OFF_HS + (size_t)NTOK * HDIM * 2;          // N*K*H bf16
constexpr size_t OFF_YE   = OFF_HE + (size_t)NTOK * KTOP * HDIM * 2;   // N*K*D bf16
constexpr size_t OFF_XB   = OFF_YE + (size_t)NTOK * KTOP * DDIM * 2;   // N*D bf16
constexpr size_t OFF_WS1T = OFF_XB + (size_t)NTOK * DDIM * 2;          // [H][D] bf16
constexpr size_t OFF_WS2T = OFF_WS1T + (size_t)DDIM * HDIM * 2;        // [D][H] bf16
constexpr size_t OFF_WE1T = OFF_WS2T + (size_t)HDIM * DDIM * 2;        // E*[H][D] bf16
constexpr size_t OFF_WE2T = OFF_WE1T + (size_t)NEXP * DDIM * HDIM * 2; // E*[D][H] bf16

// ---------------------------------------------------------------------------
// Kernel 0: zero the per-expert counters
// ---------------------------------------------------------------------------
__global__ void zero_cnt_kernel(int* cnt) {
    if (threadIdx.x < NEXP) cnt[threadIdx.x] = 0;
}

// ---------------------------------------------------------------------------
// Kernel: x (fp32) -> Xb (bf16), elementwise (float4 -> 4 bf16)
// ---------------------------------------------------------------------------
__global__ __launch_bounds__(256) void x2bf_kernel(const float* __restrict__ x,
                                                   unsigned short* __restrict__ xb) {
    int g = blockIdx.x * 256 + threadIdx.x;          // float4 group id
    float4 q = *(const float4*)(x + (size_t)g * 4);
    unsigned short h[4] = { f2bf(q.x), f2bf(q.y), f2bf(q.z), f2bf(q.w) };
    *(uint2*)(xb + (size_t)g * 4) = *(const uint2*)h;
}

// ---------------------------------------------------------------------------
// Kernel: transpose + convert weights.  src: nmat matrices [1024][1024] fp32
// row-major (k-major).  dst: bf16 [n][k] (k contiguous) per matrix.
// 64x64 tiles through LDS; coalesced loads and stores.
// ---------------------------------------------------------------------------
__global__ __launch_bounds__(256) void transpose_w_kernel(
    const float* __restrict__ W, unsigned short* __restrict__ WT) {
    const int z  = blockIdx.z;
    const float* src = W + (size_t)z * DDIM * HDIM;
    unsigned short* dst = WT + (size_t)z * DDIM * HDIM;
    const int k0 = blockIdx.x * 64;
    const int n0 = blockIdx.y * 64;

    __shared__ float tile[64][68];
    const int t = threadIdx.x;
#pragma unroll
    for (int i = 0; i < 4; ++i) {
        int lin = t + i * 256;        // 0..1023 float4 groups
        int kk  = lin >> 4;           // 16 groups per row of 64
        int c4  = lin & 15;
        float4 q = *(const float4*)(src + (size_t)(k0 + kk) * 1024 + n0 + c4 * 4);
        tile[kk][c4 * 4 + 0] = q.x; tile[kk][c4 * 4 + 1] = q.y;
        tile[kk][c4 * 4 + 2] = q.z; tile[kk][c4 * 4 + 3] = q.w;
    }
    __syncthreads();
#pragma unroll
    for (int i = 0; i < 4; ++i) {
        int lin = t + i * 256;
        int nn  = lin >> 4;           // output row (n)
        int kg  = lin & 15;           // 4-wide k group
        unsigned short h[4];
#pragma unroll
        for (int j = 0; j < 4; ++j) h[j] = f2bf(tile[kg * 4 + j][nn]);
        *(uint2*)(dst + (size_t)(n0 + nn) * 1024 + k0 + kg * 4) = *(const uint2*)h;
    }
}

// ---------------------------------------------------------------------------
// Kernel 1: router (one wave32 per token) -> top-3 weights + expert lists
// ---------------------------------------------------------------------------
__global__ __launch_bounds__(256) void router_kernel(
    const float* __restrict__ x, const float* __restrict__ rW,
    const float* __restrict__ rb, float* __restrict__ tk_w,
    int* __restrict__ list, int* __restrict__ cnt)
{
    const int wave = threadIdx.x >> 5;
    const int lane = threadIdx.x & 31;
    const int n = blockIdx.x * 8 + wave;

    float acc[NEXP];
#pragma unroll
    for (int e = 0; e < NEXP; ++e) acc[e] = 0.0f;

    for (int j = 0; j < DDIM; j += 32) {
        float xv = x[(size_t)n * DDIM + j + lane];
        const float* wrow = rW + (size_t)(j + lane) * NEXP;
#pragma unroll
        for (int e = 0; e < NEXP; ++e) acc[e] = fmaf(xv, wrow[e], acc[e]);
    }
#pragma unroll
    for (int e = 0; e < NEXP; ++e) {
        for (int off = 16; off > 0; off >>= 1)
            acc[e] += __shfl_xor(acc[e], off, 32);
    }

    if (lane == 0) {
        float mx = -1e30f;
#pragma unroll
        for (int e = 0; e < NEXP; ++e) { acc[e] += rb[e]; mx = fmaxf(mx, acc[e]); }
        float p[NEXP], s = 0.0f;
#pragma unroll
        for (int e = 0; e < NEXP; ++e) { p[e] = __expf(acc[e] - mx); s += p[e]; }
        const float inv = 1.0f / s;

        unsigned used = 0;
#pragma unroll
        for (int k = 0; k < KTOP; ++k) {
            int be = 0; float bp = -1.0f;
#pragma unroll
            for (int e = 0; e < NEXP; ++e)
                if (!((used >> e) & 1u) && p[e] > bp) { bp = p[e]; be = e; }
            used |= 1u << be;
            const int slot = n * KTOP + k;
            tk_w[slot] = bp * inv;
            int pos = atomicAdd(&cnt[be], 1);
            list[be * NTOK + pos] = slot;
        }
    }
}

// ---------------------------------------------------------------------------
// Unified WMMA GEMM: C[M x 1024] = op(A[M x 1024] @ B + bias)
//   A: bf16 [row][k];  B: pre-transposed bf16 [n][k] (k contiguous)
//   128x128 block tile, 8 waves of 32x64, K-step 32.
//   Dense tiles come in via TDM (tensor_load_to_lds + s_wait_tensorcnt);
//   gathered A rows (expert GEMMs) use per-lane bf16 loads.
// ---------------------------------------------------------------------------
template<bool GATHER, bool DIV3, bool RELU, bool OUT_BF16>
__global__ __launch_bounds__(256) void gemm_wmma(
    const unsigned short* __restrict__ Abf, const unsigned short* __restrict__ BT,
    const float* __restrict__ bias, void* __restrict__ Cout,
    const int* __restrict__ list, const int* __restrict__ cnt)
{
    const int n0 = blockIdx.x * 128;
    const int m0 = blockIdx.y * 128;

    int e = 0, Me = NTOK;
    const int* elist = nullptr;
    if (GATHER) {
        e = blockIdx.z;
        Me = cnt[e];
        if (m0 >= Me) return;
        elist = list + (size_t)e * NTOK;
    }
    const unsigned short* Bq = BT + (GATHER ? (size_t)e * DDIM * HDIM : 0);
    const float* bq = bias + (GATHER ? (size_t)e * 1024 : 0);

    __shared__ __align__(16) unsigned short As[128 * 40];  // [row][k], pad 8
    __shared__ __align__(16) unsigned short Bs[128 * 40];  // [n][k],   pad 8
    __shared__ int rows_tile[128];

    const int t = threadIdx.x;
    if (GATHER && t < 128) {
        int m = m0 + t;
        rows_tile[t] = (m < Me) ? elist[m] : elist[m0];    // safe fallback
    }
    __syncthreads();

    const unsigned asOff = (unsigned)(size_t)(void*)&As[0];
    const unsigned bsOff = (unsigned)(size_t)(void*)&Bs[0];

    const int lane = t & 31;
    const int w    = t >> 5;
    const int wm   = w & 3;    // M quadrant: rows wm*32
    const int wn   = w >> 2;   // N half:    cols wn*64

    v8f zero;
#pragma unroll
    for (int i = 0; i < 8; ++i) zero[i] = 0.0f;
    v8f acc[2][4];
#pragma unroll
    for (int mi = 0; mi < 2; ++mi)
#pragma unroll
        for (int ni = 0; ni < 4; ++ni) acc[mi][ni] = zero;

    for (int k0 = 0; k0 < 1024; k0 += 32) {
        if (GATHER) {
#if HAS_TDM
            if (w == 0) tdm_load_tile(bsOff, Bq + (size_t)n0 * 1024 + k0);
#endif
            // gathered A rows: per-lane bf16 loads
#pragma unroll
            for (int i = 0; i < 4; ++i) {
                int lin = t + i * 256;      // 0..1023 uint2 groups
                int r   = lin >> 3;
                int kg  = lin & 7;
                int slot = rows_tile[r];
                int arow = DIV3 ? (slot / 3) : slot;
                uint2 q = *(const uint2*)(Abf + (size_t)arow * 1024 + k0 + kg * 4);
                *(uint2*)(&As[r * 40 + kg * 4]) = q;
            }
#if HAS_TDM
            if (w == 0) wait_tensorcnt0();
#else
#pragma unroll
            for (int i = 0; i < 4; ++i) {
                int lin = t + i * 256;
                int r   = lin >> 3;
                int kg  = lin & 7;
                uint2 q = *(const uint2*)(Bq + (size_t)(n0 + r) * 1024 + k0 + kg * 4);
                *(uint2*)(&Bs[r * 40 + kg * 4]) = q;
            }
#endif
        } else {
#if HAS_TDM
            if (w == 0) {
                tdm_load_tile(asOff, Abf + (size_t)m0 * 1024 + k0);
                tdm_load_tile(bsOff, Bq + (size_t)n0 * 1024 + k0);
                wait_tensorcnt0();
            }
#else
#pragma unroll
            for (int i = 0; i < 4; ++i) {
                int lin = t + i * 256;
                int r   = lin >> 3;
                int kg  = lin & 7;
                uint2 qa = *(const uint2*)(Abf + (size_t)(m0 + r) * 1024 + k0 + kg * 4);
                *(uint2*)(&As[r * 40 + kg * 4]) = qa;
                uint2 qb = *(const uint2*)(Bq + (size_t)(n0 + r) * 1024 + k0 + kg * 4);
                *(uint2*)(&Bs[r * 40 + kg * 4]) = qb;
            }
#endif
        }
        __syncthreads();

        // ---- fragments per ISA 16-bit layouts, then 2x4 WMMA ----
        FragBF afr[2];
#pragma unroll
        for (int mi = 0; mi < 2; ++mi) {
            int row = wm * 32 + mi * 16 + (lane & 15);
            int kb  = (lane >> 4) * 8;      // lanes 16-31 hold K+8 / K+24
            afr[mi].q[0] = *(const v4u*)(&As[row * 40 + kb]);
            afr[mi].q[1] = *(const v4u*)(&As[row * 40 + 16 + kb]);
        }
#pragma unroll
        for (int ni = 0; ni < 4; ++ni) {
            int col = wn * 64 + ni * 16 + (lane & 15);
            int kb  = (lane >> 4) * 8;      // lanes 16-31 hold K+8 / K+24
            FragBF bfr;
            bfr.q[0] = *(const v4u*)(&Bs[col * 40 + kb]);
            bfr.q[1] = *(const v4u*)(&Bs[col * 40 + 16 + kb]);
#pragma unroll
            for (int mi = 0; mi < 2; ++mi) {
                acc[mi][ni] = __builtin_amdgcn_wmma_f32_16x16x32_bf16(
                    false, afr[mi].v, false, bfr.v,
                    (short)0, acc[mi][ni], false, false);
            }
        }
        __syncthreads();
    }

    // ---- epilogue: bias (+ReLU), scatter per C-matrix VGPR layout ----
#pragma unroll
    for (int mi = 0; mi < 2; ++mi) {
#pragma unroll
        for (int ni = 0; ni < 4; ++ni) {
            int col = n0 + wn * 64 + ni * 16 + (lane & 15);
            float bv = bq[col];
#pragma unroll
            for (int r = 0; r < 8; ++r) {
                int rloc = wm * 32 + mi * 16 + (lane >> 4) * 8 + r;
                float v = acc[mi][ni][r] + bv;
                if (RELU) v = fmaxf(v, 0.0f);
                int orow; bool ok = true;
                if (GATHER) {
                    ok = (m0 + rloc) < Me;
                    orow = rows_tile[rloc];
                } else {
                    orow = m0 + rloc;
                }
                if (ok) {
                    if (OUT_BF16)
                        ((unsigned short*)Cout)[(size_t)orow * 1024 + col] = f2bf(v);
                    else
                        ((float*)Cout)[(size_t)orow * 1024 + col] = v;
                }
            }
        }
    }
}

// ---------------------------------------------------------------------------
// Kernel: out[n,d] += sum_k tk_w[n*3+k] * Ye[n*3+k][d]
// ---------------------------------------------------------------------------
__global__ __launch_bounds__(256) void combine_kernel(
    float* __restrict__ out, const unsigned short* __restrict__ Ye,
    const float* __restrict__ tk_w)
{
    int idx = blockIdx.x * 256 + threadIdx.x;
    int n = idx >> 10;
    int d = idx & 1023;
    float s = out[idx];
#pragma unroll
    for (int k = 0; k < KTOP; ++k) {
        int slot = n * KTOP + k;
        s += tk_w[slot] * bf2f(Ye[(size_t)slot * 1024 + d]);
    }
    out[idx] = s;
}

// ---------------------------------------------------------------------------
extern "C" void kernel_launch(void* const* d_in, const int* in_sizes, int n_in,
                              void* d_out, int out_size, void* d_ws, size_t ws_size,
                              hipStream_t stream) {
    const float* x   = (const float*)d_in[0];
    const float* rW  = (const float*)d_in[1];
    const float* rb  = (const float*)d_in[2];
    const float* sW1 = (const float*)d_in[3];
    const float* sb1 = (const float*)d_in[4];
    const float* sW2 = (const float*)d_in[5];
    const float* sb2 = (const float*)d_in[6];
    const float* eW1 = (const float*)d_in[7];
    const float* eb1 = (const float*)d_in[8];
    const float* eW2 = (const float*)d_in[9];
    const float* eb2 = (const float*)d_in[10];
    float* out = (float*)d_out;

    char* ws = (char*)d_ws;
    int*            cnt  = (int*)(ws + OFF_CNT);
    int*            list = (int*)(ws + OFF_LIST);
    float*          tkw  = (float*)(ws + OFF_TKW);
    unsigned short* Hs   = (unsigned short*)(ws + OFF_HS);
    unsigned short* He   = (unsigned short*)(ws + OFF_HE);
    unsigned short* Ye   = (unsigned short*)(ws + OFF_YE);
    unsigned short* Xb   = (unsigned short*)(ws + OFF_XB);
    unsigned short* Ws1T = (unsigned short*)(ws + OFF_WS1T);
    unsigned short* Ws2T = (unsigned short*)(ws + OFF_WS2T);
    unsigned short* We1T = (unsigned short*)(ws + OFF_WE1T);
    unsigned short* We2T = (unsigned short*)(ws + OFF_WE2T);

    // 0) reset expert counters; convert x; transpose+convert all weights
    zero_cnt_kernel<<<1, 32, 0, stream>>>(cnt);
    x2bf_kernel<<<(NTOK * DDIM) / (256 * 4), 256, 0, stream>>>(x, Xb);
    transpose_w_kernel<<<dim3(16, 16, 1),    256, 0, stream>>>(sW1, Ws1T);
    transpose_w_kernel<<<dim3(16, 16, 1),    256, 0, stream>>>(sW2, Ws2T);
    transpose_w_kernel<<<dim3(16, 16, NEXP), 256, 0, stream>>>(eW1, We1T);
    transpose_w_kernel<<<dim3(16, 16, NEXP), 256, 0, stream>>>(eW2, We2T);

    // 1) router: softmax + top-3 + compaction lists
    router_kernel<<<NTOK / 8, 256, 0, stream>>>(x, rW, rb, tkw, list, cnt);

    // 2) shared FFN stage 1: Hs = relu(x @ sW1 + sb1)   (bf16)
    gemm_wmma<false, false, true, true>
        <<<dim3(HDIM / 128, NTOK / 128), 256, 0, stream>>>(
            Xb, Ws1T, sb1, Hs, nullptr, nullptr);

    // 3) routed stage 1 (compacted): He[slot] = relu(x @ eW1[e] + eb1[e])
    gemm_wmma<true, true, true, true>
        <<<dim3(HDIM / 128, NTOK / 128, NEXP), 256, 0, stream>>>(
            Xb, We1T, eb1, He, list, cnt);

    // 4) shared FFN stage 2: out = Hs @ sW2 + sb2       (fp32)
    gemm_wmma<false, false, false, false>
        <<<dim3(DDIM / 128, NTOK / 128), 256, 0, stream>>>(
            Hs, Ws2T, sb2, out, nullptr, nullptr);

    // 5) routed stage 2: Ye[slot] = He[slot] @ eW2[e] + eb2[e]  (bf16)
    gemm_wmma<true, false, false, true>
        <<<dim3(DDIM / 128, NTOK / 128, NEXP), 256, 0, stream>>>(
            He, We2T, eb2, Ye, list, cnt);

    // 6) combine: out += sum_k w_k * Ye[n*3+k]
    combine_kernel<<<(NTOK * DDIM) / 256, 256, 0, stream>>>(out, Ye, tkw);
}
